// SpatialAttentionBlock_55499567399249
// MI455X (gfx1250) — matrix operand order repaired and checked by reading, not compiled, not run on hardware
//
#include <hip/hip_runtime.h>
#include <hip/hip_bf16.h>

#define SEQL    2048
#define DMODEL  1024
#define NHEAD   16
#define HDIM    64
#define QB      128      // query rows per flash block
#define TKTILE  128      // key/value tile
#define TKP     136      // padded f16 row stride for Vt/P LDS tiles (16B-aligned rows)
#define KLD     72       // padded f16 row stride for K LDS tile (144B rows, conflict-free)

typedef __attribute__((ext_vector_type(16))) _Float16 v16h;
typedef __attribute__((ext_vector_type(8)))  _Float16 v8h;
typedef __attribute__((ext_vector_type(8)))  float    v8f;

// ---------------------------------------------------------------------------
// WMMA helpers (v_wmma_f32_16x16x32_f16, f32 accumulate)
// ---------------------------------------------------------------------------
__device__ __forceinline__ v8f wmma_f16(v16h a, v16h b, v8f c) {
  return __builtin_amdgcn_wmma_f32_16x16x32_f16(
      /*neg_a=*/false, a, /*neg_b=*/false, b,
      /*c_mod=*/(short)0, c, /*reuse_a=*/false, /*reuse_b=*/false);
}

// A-fragment (16x32, M x K) from row-major [rows][ld] f16.
// Lane (m = lane&15, g = lane>>4) holds K = {k0+8g .. +7} and {k0+16+8g .. +7}:
// two contiguous 16-byte chunks -> 2x global_load_b128 / ds_load_b128.
__device__ __forceinline__ v16h load_a_frag(const _Float16* base, int ld, int k0, int lane) {
  const int m = lane & 15, g = lane >> 4;
  const _Float16* p = base + (size_t)m * ld + (k0 + 8 * g);
  v8h lo = *(const v8h*)p;
  v8h hi = *(const v8h*)(p + 16);
  v16h out;
#pragma unroll
  for (int i = 0; i < 8; ++i) { out[i] = lo[i]; out[i + 8] = hi[i]; }
  return out;
}

// B-fragment for B[k][n] = W[n][k] where W is row-major [rows][ld] f16
// (i.e. GEMM against W^T). Lane (n = lane&15, g = lane>>4) holds
// K = {k0+8g .. +7} and {k0+16+8g .. +7}: two 16-byte chunks of row n.
__device__ __forceinline__ v16h load_b_fragT(const _Float16* base, int ld, int k0, int lane) {
  const int n = lane & 15, g = lane >> 4;
  const _Float16* p = base + (size_t)n * ld + (k0 + 8 * g);
  v8h lo = *(const v8h*)p;
  v8h hi = *(const v8h*)(p + 16);
  v16h out;
#pragma unroll
  for (int i = 0; i < 8; ++i) { out[i] = lo[i]; out[i + 8] = hi[i]; }
  return out;
}

// CDNA5 async global->LDS copy (16B per lane, tracked by ASYNCcnt).
__device__ __forceinline__ void async_ld_b128(unsigned lds_off, const void* gaddr) {
  asm volatile("global_load_async_to_lds_b128 %0, %1, off"
               :: "v"(lds_off), "v"(gaddr) : "memory");
}
__device__ __forceinline__ void wait_async0() {
  asm volatile("s_wait_asynccnt 0x0" ::: "memory");
}

// 16-lane (half-wave) row reductions; xor masks 1,2,4,8 stay within the group.
__device__ __forceinline__ float hgrp_max(float v) {
  v = fmaxf(v, __shfl_xor(v, 1, 32));
  v = fmaxf(v, __shfl_xor(v, 2, 32));
  v = fmaxf(v, __shfl_xor(v, 4, 32));
  v = fmaxf(v, __shfl_xor(v, 8, 32));
  return v;
}
__device__ __forceinline__ float hgrp_sum(float v) {
  v += __shfl_xor(v, 1, 32);
  v += __shfl_xor(v, 2, 32);
  v += __shfl_xor(v, 4, 32);
  v += __shfl_xor(v, 8, 32);
  return v;
}

// ---------------------------------------------------------------------------
// f32 -> f16 convert
// ---------------------------------------------------------------------------
__global__ __launch_bounds__(256) void cvt_kernel(const float* __restrict__ x,
                                                  _Float16* __restrict__ y, int n4) {
  int i = blockIdx.x * blockDim.x + threadIdx.x;
  int stride = gridDim.x * blockDim.x;
  for (; i < n4; i += stride) {
    float4 v = ((const float4*)x)[i];
    y[4 * i + 0] = (_Float16)v.x;
    y[4 * i + 1] = (_Float16)v.y;
    y[4 * i + 2] = (_Float16)v.z;
    y[4 * i + 3] = (_Float16)v.w;
  }
}

// ---------------------------------------------------------------------------
// OUT[m][n] = sum_k A[m][k] * W[n][k] + bias[n], optional fused RoPE.
// Wave tile: 16 rows x 64 cols (one head wide => RoPE pair (d, d+32) is
// lane-local between accumulator subtiles j and j+2). K-loop software-pipelined.
// ---------------------------------------------------------------------------
template <bool ROPE, bool OUTF32>
__global__ __launch_bounds__(256) void gemm_xwT_kernel(
    const _Float16* __restrict__ A, const _Float16* __restrict__ W,
    const float* __restrict__ bias, void* __restrict__ outp,
    int M, int N, int Kd) {
  const int tid = threadIdx.x;
  const int wave = tid >> 5, lane = tid & 31;
  const int g = lane >> 4, ln = lane & 15;
  const int gw = blockIdx.x * 8 + wave;
  const int tilesN = N / 64;
  const int row0 = (gw / tilesN) * 16;
  const int col0 = (gw % tilesN) * 64;
  if (row0 >= M) return;

  v8f acc[4] = {};
  const _Float16* Abase = A + (size_t)row0 * Kd;

  v16h a_cur = load_a_frag(Abase, Kd, 0, lane);
  v16h b_cur[4];
#pragma unroll
  for (int j = 0; j < 4; ++j)
    b_cur[j] = load_b_fragT(W + (size_t)(col0 + 16 * j) * Kd, Kd, 0, lane);

  for (int k0 = 0; k0 < Kd; k0 += 32) {
    const int kn = (k0 + 32 < Kd) ? (k0 + 32) : 0;  // wrap: unconditional loads
    v16h a_nxt = load_a_frag(Abase, Kd, kn, lane);
    v16h b_nxt[4];
#pragma unroll
    for (int j = 0; j < 4; ++j)
      b_nxt[j] = load_b_fragT(W + (size_t)(col0 + 16 * j) * Kd, Kd, kn, lane);
#pragma unroll
    for (int j = 0; j < 4; ++j)
      acc[j] = wmma_f16(a_cur, b_cur[j], acc[j]);
    a_cur = a_nxt;
#pragma unroll
    for (int j = 0; j < 4; ++j) b_cur[j] = b_nxt[j];
  }

  float bj[4];
#pragma unroll
  for (int j = 0; j < 4; ++j) bj[j] = bias[col0 + 16 * j + ln];

  if (ROPE) {
    // eqx half-split RoPE: pair (d, d+32), freq = theta^(-d/32), angle = s*freq
#pragma unroll
    for (int j = 0; j < 2; ++j) {
      const int d = 16 * j + ln;  // 0..31
      const float freq = __expf(-0.28782313662425572f * (float)d);  // ln(1e4)/32
#pragma unroll
      for (int r = 0; r < 8; ++r) {
        const int s = row0 + r + 8 * g;
        float sn, cs;
        __sincosf((float)s * freq, &sn, &cs);
        const float x1 = acc[j][r] + bj[j];
        const float x2 = acc[j + 2][r] + bj[j + 2];
        acc[j][r]     = x1 * cs - x2 * sn;
        acc[j + 2][r] = x2 * cs + x1 * sn;
      }
    }
  } else {
#pragma unroll
    for (int j = 0; j < 4; ++j)
#pragma unroll
      for (int r = 0; r < 8; ++r) acc[j][r] += bj[j];
  }

  if (OUTF32) {
    float* O = (float*)outp;
#pragma unroll
    for (int j = 0; j < 4; ++j)
#pragma unroll
      for (int r = 0; r < 8; ++r)
        O[(size_t)(row0 + r + 8 * g) * N + col0 + 16 * j + ln] = acc[j][r];
  } else {
    _Float16* O = (_Float16*)outp;
#pragma unroll
    for (int j = 0; j < 4; ++j)
#pragma unroll
      for (int r = 0; r < 8; ++r)
        O[(size_t)(row0 + r + 8 * g) * N + col0 + 16 * j + ln] = (_Float16)acc[j][r];
  }
}

// ---------------------------------------------------------------------------
// Flash attention: block = (head, 128-query tile); 8 waves x 16 query rows.
// K tile staged via async global->LDS (ASYNCcnt), V tile transposed into LDS.
// P staging buffer aliases the K tile (phase barrier separates reads/writes).
// Online softmax with 16-lane shuffle reductions; all WMMA operands in LDS
// during the hot loop so per-WMMA waits are ds-latency, not L2-latency.
// ---------------------------------------------------------------------------
__global__ __launch_bounds__(256) void flash_attn_kernel(
    const _Float16* __restrict__ Q, const _Float16* __restrict__ K,
    const _Float16* __restrict__ V, _Float16* __restrict__ O) {
  __shared__ _Float16 Vt[HDIM][TKP];        // V^T tile: Vt[d][t]      (17408 B)
  __shared__ _Float16 PK[8][16][TKP];       // P staging / K tile alias (34816 B)
  _Float16* Kl = &PK[0][0][0];              // Kl[t*KLD + d], 128*72*2 = 18432 B

  const int h  = blockIdx.x & (NHEAD - 1);
  const int qb = blockIdx.x / NHEAD;
  const int tid = threadIdx.x;
  const int wave = tid >> 5, lane = tid & 31;
  const int g = lane >> 4, ln = lane & 15;
  const int row0 = qb * QB + wave * 16;

  // Q fragments for this wave's 16 rows (head dim 64 = 2 K-steps), loaded once.
  const _Float16* Qb = Q + (size_t)row0 * DMODEL + h * HDIM;
  const v16h qa0 = load_a_frag(Qb, DMODEL, 0, lane);
  const v16h qa1 = load_a_frag(Qb, DMODEL, 32, lane);

  float mrun[8], lrun[8];
  v8f acc[4] = {};
#pragma unroll
  for (int r = 0; r < 8; ++r) { mrun[r] = -1e30f; lrun[r] = 0.0f; }

  for (int t0 = 0; t0 < SEQL; t0 += TKTILE) {
    __syncthreads();  // previous iteration done with Vt / PK

    // K tile: straight copy via CDNA5 async global->LDS, 16B per lane.
    for (int idx = tid; idx < TKTILE * (HDIM / 8); idx += 256) {
      const int t = idx >> 3;
      const int c = (idx & 7) * 8;
      const unsigned loff = (unsigned)(uintptr_t)(&Kl[t * KLD + c]);
      async_ld_b128(loff, (const void*)(K + (size_t)(t0 + t) * DMODEL + h * HDIM + c));
    }
    // V tile: transpose-stage (b128 global loads, f16 LDS scatter).
    for (int idx = tid; idx < TKTILE * (HDIM / 8); idx += 256) {
      const int t = idx >> 3;
      const int dg = (idx & 7) * 8;
      v8h vv = *(const v8h*)(V + (size_t)(t0 + t) * DMODEL + h * HDIM + dg);
#pragma unroll
      for (int u = 0; u < 8; ++u) Vt[dg + u][t] = vv[u];
    }
    wait_async0();
    __syncthreads();

    // Logits: 16 queries x 128 keys from LDS, K-dim 64 -> 8 n-tiles x 2 WMMAs.
    v8f S[8];
#pragma unroll
    for (int tt = 0; tt < 8; ++tt) {
      const _Float16* Kp = Kl + (size_t)(tt * 16) * KLD;
      v16h b0 = load_b_fragT(Kp, KLD, 0, lane);
      v16h b1 = load_b_fragT(Kp, KLD, 32, lane);
      v8f c = {};
      c = wmma_f16(qa0, b0, c);
      c = wmma_f16(qa1, b1, c);
      S[tt] = c * 0.125f;  // 1/sqrt(64)
    }

    // Online softmax update (row stats replicated across each 16-lane group).
    float corr[8];
#pragma unroll
    for (int r = 0; r < 8; ++r) {
      float v = S[0][r];
#pragma unroll
      for (int tt = 1; tt < 8; ++tt) v = fmaxf(v, S[tt][r]);
      v = hgrp_max(v);
      const float mnew = fmaxf(mrun[r], v);
      corr[r] = __expf(mrun[r] - mnew);
      mrun[r] = mnew;
    }
#pragma unroll
    for (int tt = 0; tt < 8; ++tt)
#pragma unroll
      for (int r = 0; r < 8; ++r) S[tt][r] = __expf(S[tt][r] - mrun[r]);
#pragma unroll
    for (int r = 0; r < 8; ++r) {
      float s = S[0][r];
#pragma unroll
      for (int tt = 1; tt < 8; ++tt) s += S[tt][r];
      lrun[r] = lrun[r] * corr[r] + hgrp_sum(s);
#pragma unroll
      for (int j = 0; j < 4; ++j) acc[j][r] *= corr[r];
    }

    __syncthreads();  // all waves done reading Kl before P overwrites it

    // D-layout f32 -> A-layout f16 via per-wave LDS (LDS is in-order per wave).
#pragma unroll
    for (int tt = 0; tt < 8; ++tt)
#pragma unroll
      for (int r = 0; r < 8; ++r)
        PK[wave][r + 8 * g][tt * 16 + ln] = (_Float16)S[tt][r];
    asm volatile("s_wait_dscnt 0x0" ::: "memory");

    // PV: O += P(16x128) @ V(128x64): 4 K-steps x 4 d-tiles (LDS fragments).
#pragma unroll
    for (int kk = 0; kk < 4; ++kk) {
      v16h pa = load_a_frag(&PK[wave][0][0], TKP, kk * 32, lane);
#pragma unroll
      for (int j = 0; j < 4; ++j) {
        v16h vb = load_b_fragT(&Vt[j * 16][0], TKP, kk * 32, lane);
        acc[j] = wmma_f16(pa, vb, acc[j]);
      }
    }
  }

  // Normalize and store O as f16 [s][h*64+d].
#pragma unroll
  for (int r = 0; r < 8; ++r) {
    const float inv = 1.0f / lrun[r];
    const size_t base = (size_t)(row0 + r + 8 * g) * DMODEL + h * HDIM;
#pragma unroll
    for (int j = 0; j < 4; ++j)
      O[base + j * 16 + ln] = (_Float16)(acc[j][r] * inv);
  }
}

// ---------------------------------------------------------------------------
// Launch
// ---------------------------------------------------------------------------
extern "C" void kernel_launch(void* const* d_in, const int* in_sizes, int n_in,
                              void* d_out, int out_size, void* d_ws, size_t ws_size,
                              hipStream_t stream) {
  const float* inX = (const float*)d_in[0];
  const float* Wq  = (const float*)d_in[1];
  const float* bq  = (const float*)d_in[2];
  const float* Wk  = (const float*)d_in[3];
  const float* bk  = (const float*)d_in[4];
  const float* Wv  = (const float*)d_in[5];
  const float* bv  = (const float*)d_in[6];
  const float* Wo  = (const float*)d_in[7];
  const float* bo  = (const float*)d_in[8];

  // Workspace layout (f16), total ~28MB — resident in 192MB L2.
  char* ws = (char*)d_ws;
  const size_t XW = (size_t)SEQL * DMODEL;    // 2M elems
  const size_t WW = (size_t)DMODEL * DMODEL;  // 1M elems
  _Float16* Xh  = (_Float16*)(ws);
  _Float16* Wqh = (_Float16*)(ws + 2 * XW);
  _Float16* Wkh = (_Float16*)(ws + 2 * XW + 2 * WW);
  _Float16* Wvh = (_Float16*)(ws + 2 * XW + 4 * WW);
  _Float16* Woh = (_Float16*)(ws + 2 * XW + 6 * WW);
  _Float16* Qh  = (_Float16*)(ws + 2 * XW + 8 * WW);
  _Float16* Kh  = (_Float16*)(ws + 4 * XW + 8 * WW);
  _Float16* Vh  = (_Float16*)(ws + 6 * XW + 8 * WW);
  _Float16* Oh  = (_Float16*)(ws + 8 * XW + 8 * WW);

  // 1. f32 -> f16 conversions
  cvt_kernel<<<1024, 256, 0, stream>>>(inX, Xh, (int)(XW / 4));
  cvt_kernel<<<512,  256, 0, stream>>>(Wq, Wqh, (int)(WW / 4));
  cvt_kernel<<<512,  256, 0, stream>>>(Wk, Wkh, (int)(WW / 4));
  cvt_kernel<<<512,  256, 0, stream>>>(Wv, Wvh, (int)(WW / 4));
  cvt_kernel<<<512,  256, 0, stream>>>(Wo, Woh, (int)(WW / 4));

  // 2. QKV projections (+bias, RoPE on Q,K): 256 blocks each
  const int gemmBlocks = (SEQL / 16) * (DMODEL / 64) / 8;
  gemm_xwT_kernel<true,  false><<<gemmBlocks, 256, 0, stream>>>(Xh, Wqh, bq, Qh, SEQL, DMODEL, DMODEL);
  gemm_xwT_kernel<true,  false><<<gemmBlocks, 256, 0, stream>>>(Xh, Wkh, bk, Kh, SEQL, DMODEL, DMODEL);
  gemm_xwT_kernel<false, false><<<gemmBlocks, 256, 0, stream>>>(Xh, Wvh, bv, Vh, SEQL, DMODEL, DMODEL);

  // 3. Fused flash attention: 16 heads x 16 query tiles
  flash_attn_kernel<<<NHEAD * (SEQL / QB), 256, 0, stream>>>(Qh, Kh, Vh, Oh);

  // 4. Output projection -> f32 d_out
  gemm_xwT_kernel<false, true><<<gemmBlocks, 256, 0, stream>>>(Oh, Woh, bo, d_out, SEQL, DMODEL, DMODEL);
}
